// LocalPatchAttention_43499428774067
// MI455X (gfx1250) — matrix operand order repaired and checked by reading, not compiled, not run on hardware
//
#include <hip/hip_runtime.h>
#include <hip/hip_bf16.h>

typedef __attribute__((ext_vector_type(16))) __bf16    v16bf;
typedef __attribute__((ext_vector_type(8)))  float     v8f;
typedef __attribute__((ext_vector_type(4)))  unsigned  u32x4;

union BFv {
    v16bf v;
    unsigned short u[16];
    u32x4 q[2];
};

__device__ __forceinline__ unsigned short f2bf(float x) {
    return __builtin_bit_cast(unsigned short, (__bf16)x);   // native cvt, RTNE
}

__device__ __forceinline__ v8f wmma_bf16(const BFv& a, const BFv& b, v8f c) {
    return __builtin_amdgcn_wmma_f32_16x16x32_bf16(
        false, a.v, false, b.v, (short)0, c, false, false);
}

// workspace layout (elements of unsigned short)
#define SPAT_ELEMS  (32768u * 2048u)         // bf16 intermediate, 128 MiB
#define WTAP_ELEMS  (9 * 64 * 128)           // conv weights, tap-major bf16
#define QW_ELEMS    (64 * 64)
#define K_ELEMS     (128 * 64)

// ---------------------------------------------------------------------------
// Prep: one-shot fp32 -> bf16 weight conversion into workspace
// ---------------------------------------------------------------------------
__global__ __launch_bounds__(256) void prep_weights(
    const float* __restrict__ cW, const float* __restrict__ qW,
    const float* __restrict__ Km,
    unsigned short* __restrict__ wtap,   // [9][64][128]
    unsigned short* __restrict__ qWbf,   // [64][64]
    unsigned short* __restrict__ Kbf)    // [128][64]
{
    int i = blockIdx.x * 256 + threadIdx.x;
    if (i < WTAP_ELEMS) {
        int tap = i / 8192, r = i - tap * 8192;      // r = co*128+ci
        wtap[i] = f2bf(cW[(size_t)r * 9 + tap]);
    }
    if (i < QW_ELEMS) qWbf[i] = f2bf(qW[i]);
    if (i < K_ELEMS)  Kbf[i]  = f2bf(Km[i]);
}

// ---------------------------------------------------------------------------
// Stage 1: patch attention.  One wave per (b,hy,wx) location; the 4 waves of
// a block own 4 consecutive wx -> cooperative (coalesced) q/v loads.
// Writes spat as bf16 in layout [b][hy][wx][p(16)][c(128)]  (256B/pixel rows)
// ---------------------------------------------------------------------------
#define WAVES1 4
__global__ __launch_bounds__(128) void attn_stage1(
    const float* __restrict__ qg,  const float* __restrict__ vg,
    const unsigned short* __restrict__ qWbf, const float* __restrict__ qb,
    const float* __restrict__ vW,  const float* __restrict__ vb,
    const unsigned short* __restrict__ Kbf,
    const float* __restrict__ qn_g, const float* __restrict__ qn_b,
    const float* __restrict__ vn_g, const float* __restrict__ vn_b,
    unsigned short* __restrict__ spat)
{
    __shared__ __align__(16) unsigned short qWs[64 * 64];   // [n][k] bf16
    __shared__ __align__(16) unsigned short Ks [128 * 64];  // [n][k] bf16
    __shared__ float qbs[64], qgs[64], qos[64];
    __shared__ float vbs[128], vgs[128], vos[128];
    __shared__ __align__(16) float qtile[WAVES1][16 * 64];  // per-wave staging
    __shared__ float vns[WAVES1][128];                      // raw v -> normalized
    __shared__ float Vs [WAVES1][128];

    const int tid  = threadIdx.x;
    const int wave = tid >> 5;
    const int lane = tid & 31;
    const int m    = lane & 15;
    const int s    = lane >> 4;

    const int loc0 = blockIdx.x * WAVES1;         // 4 consecutive locations
    const int wx0  = loc0 & 63;
    const int hy   = (loc0 >> 6) & 63;
    const int b    = loc0 >> 12;

    // ---- pre-converted weights -> LDS (16B copies, no conversion) ---------
    for (int i = tid; i < 512;  i += 128) ((u32x4*)qWs)[i] = ((const u32x4*)qWbf)[i];
    for (int i = tid; i < 1024; i += 128) ((u32x4*)Ks)[i]  = ((const u32x4*)Kbf)[i];
    if (tid < 64)  { qbs[tid] = qb[tid]; qgs[tid] = qn_g[tid]; qos[tid] = qn_b[tid]; }
    if (tid < 128) { vbs[tid] = vb[tid]; vgs[tid] = vn_g[tid]; vos[tid] = vn_b[tid]; }

    // ---- cooperative q load: 64B contiguous segments across the 4 patches -
    for (int i = tid; i < 4096; i += 128) {
        int c = i >> 6, pr = (i >> 4) & 3, x16 = i & 15;
        int w4 = x16 >> 2, pc = x16 & 3;
        size_t gi = (((size_t)b * 64 + c) * 256 + (hy * 4 + pr)) * 256 + wx0 * 4 + x16;
        qtile[w4][(pr * 4 + pc) * 64 + c] = qg[gi];
    }
    // ---- cooperative v load: 4 consecutive wx per channel ------------------
    for (int i = tid; i < 512; i += 128) {
        int c = i >> 2, w4 = i & 3;
        vns[w4][c] = vg[(((size_t)b * 128 + c) * 64 + hy) * 64 + wx0 + w4];
    }
    __syncthreads();

    float* qt = qtile[wave];

    // ---- v LayerNorm + V = vn @ vW^T + vb  (VALU fp32) --------------------
    float vv[4];
    #pragma unroll
    for (int i = 0; i < 4; ++i) vv[i] = vns[wave][lane + 32 * i];
    float s1 = 0.f, s2 = 0.f;
    #pragma unroll
    for (int i = 0; i < 4; ++i) { s1 += vv[i]; s2 += vv[i] * vv[i]; }
    #pragma unroll
    for (int off = 16; off > 0; off >>= 1) {
        s1 += __shfl_xor(s1, off, 32);
        s2 += __shfl_xor(s2, off, 32);
    }
    float mu2 = s1 * (1.f / 128.f);
    float rs2 = rsqrtf(s2 * (1.f / 128.f) - mu2 * mu2 + 1e-5f);
    #pragma unroll
    for (int i = 0; i < 4; ++i) {
        int c = lane + 32 * i;
        vns[wave][c] = (vv[i] - mu2) * rs2 * vgs[c] + vos[c];
    }
    #pragma unroll
    for (int i = 0; i < 4; ++i) {
        int n = lane + 32 * i;
        const float4* wr = (const float4*)&vW[n * 128];
        float acc = vbs[n];
        for (int k4 = 0; k4 < 32; ++k4) {
            float4 w4 = wr[k4];
            acc += w4.x * vns[wave][k4 * 4 + 0] + w4.y * vns[wave][k4 * 4 + 1]
                 + w4.z * vns[wave][k4 * 4 + 2] + w4.w * vns[wave][k4 * 4 + 3];
        }
        Vs[wave][n] = acc;
    }

    // ---- q LayerNorm stats (lane owns row m; halves combined by shfl) -----
    float qs1 = 0.f, qs2 = 0.f;
    for (int c = 0; c < 32; ++c) {
        float x = qt[m * 64 + s * 32 + c];
        qs1 += x; qs2 += x * x;
    }
    qs1 += __shfl_xor(qs1, 16, 32);
    qs2 += __shfl_xor(qs2, 16, 32);
    float mu = qs1 * (1.f / 64.f);
    float rs = rsqrtf(qs2 * (1.f / 64.f) - mu * mu + 1e-5f);

    // ---- pack A (layernormed q, 16x32 bf16 per K-step) --------------------
    BFv a[2];
    #pragma unroll
    for (int kk = 0; kk < 2; ++kk)
        #pragma unroll
        for (int j = 0; j < 16; ++j) {
            int K = kk * 32 + (s << 4) + j;
            float x = (qt[m * 64 + K] - mu) * rs * qgs[K] + qos[K];
            a[kk].u[j] = f2bf(x);
        }

    // ---- Q = qn @ qW^T : 4 N-tiles x 2 K-steps ----------------------------
    v8f qacc[4];
    #pragma unroll
    for (int t = 0; t < 4; ++t) {
        v8f c = {0.f, 0.f, 0.f, 0.f, 0.f, 0.f, 0.f, 0.f};
        #pragma unroll
        for (int kk = 0; kk < 2; ++kk) {
            BFv bm;
            const unsigned short* bp = &qWs[(t * 16 + m) * 64 + kk * 32 + s * 16];
            bm.q[0] = *(const u32x4*)bp;
            bm.q[1] = *(const u32x4*)(bp + 8);
            c = wmma_bf16(a[kk], bm, c);
        }
        qacc[t] = c;
    }
    // stage Q (+qb) back into qtile
    #pragma unroll
    for (int t = 0; t < 4; ++t)
        #pragma unroll
        for (int r = 0; r < 8; ++r) {
            int row = r + s * 8, col = t * 16 + m;
            qt[row * 64 + col] = qacc[t][r] + qbs[col];
        }

    // ---- repack Q as A, logits = Q @ K^T : 8 N-tiles x 2 K-steps ----------
    BFv a2[2];
    #pragma unroll
    for (int kk = 0; kk < 2; ++kk)
        #pragma unroll
        for (int j = 0; j < 16; ++j) {
            int K = kk * 32 + (s << 4) + j;
            a2[kk].u[j] = f2bf(qt[m * 64 + K]);
        }
    v8f lg[8];
    #pragma unroll
    for (int t = 0; t < 8; ++t) {
        v8f c = {0.f, 0.f, 0.f, 0.f, 0.f, 0.f, 0.f, 0.f};
        #pragma unroll
        for (int kk = 0; kk < 2; ++kk) {
            BFv bm;
            const unsigned short* bp = &Ks[(t * 16 + m) * 64 + kk * 32 + s * 16];
            bm.q[0] = *(const u32x4*)bp;
            bm.q[1] = *(const u32x4*)(bp + 8);
            c = wmma_bf16(a2[kk], bm, c);
        }
        lg[t] = c;
    }

    // ---- x = sigmoid(logits*scale) * V  -> bf16 stage in LDS --------------
    unsigned short* xst = (unsigned short*)qt;   // reuse 4KB: [16][128] bf16
    const float scale = 0.125f;                  // Cq^-0.5
    #pragma unroll
    for (int t = 0; t < 8; ++t)
        #pragma unroll
        for (int r = 0; r < 8; ++r) {
            int row = r + s * 8, n = t * 16 + m;
            float x = lg[t][r] * scale;
            float sg = 1.f / (1.f + __expf(-x));
            xst[row * 128 + n] = f2bf(sg * Vs[wave][n]);
        }

    // ---- coalesced 4KB store of this location's patch ---------------------
    u32x4* dst = (u32x4*)&spat[(size_t)(loc0 + wave) * 2048];
    const u32x4* src = (const u32x4*)xst;
    for (int i = lane; i < 256; i += 32) dst[i] = src[i];
}

// ---------------------------------------------------------------------------
// Stage 2: 3x3 conv 128->64 (implicit GEMM) + bias + residual.
// Block: 4x8 output pixels x 64 co.  Wave tile: 16 pixels x 16 co.
// ---------------------------------------------------------------------------
__global__ __launch_bounds__(256) void conv_stage2(
    const unsigned short* __restrict__ spat,  // [b][hy][wx][p][c] bf16
    const unsigned short* __restrict__ wtap,  // [9][64][128] bf16
    const float* __restrict__ cb,
    const float* __restrict__ qin,            // residual
    float* __restrict__ out)
{
    __shared__ __align__(16) unsigned short acts[6 * 10 * 128]; // halo [r][c][ci]
    __shared__ __align__(16) unsigned short wt[64 * 128];       // per-tap [co][ci]
    __shared__ float outT[64 * 32];                              // [co][pixel]

    const int tid = threadIdx.x;
    const int bid = blockIdx.x;
    const int tx  = bid & 31;           // W/8
    const int ty  = (bid >> 5) & 63;    // H/4
    const int b   = bid >> 11;
    const int Y0  = ty * 4, X0 = tx * 8;

    // ---- load halo (6x10 pixels x 128 ci), zero-padded --------------------
    if (tid < 240) {
        int hp = tid >> 2, sub = tid & 3;  // 4 threads / pixel, 64B each
        int gy = Y0 - 1 + hp / 10;
        int gx = X0 - 1 + hp % 10;
        u32x4* d = (u32x4*)&acts[hp * 128];
        if (gy >= 0 && gy < 256 && gx >= 0 && gx < 256) {
            int hy = gy >> 2, wx = gx >> 2, p = (gy & 3) * 4 + (gx & 3);
            size_t so = ((((size_t)b * 64 + hy) * 64 + wx) * 16 + p) * 128;
            const u32x4* sp = (const u32x4*)&spat[so];
            #pragma unroll
            for (int i = 0; i < 4; ++i) d[sub * 4 + i] = sp[sub * 4 + i];
        } else {
            u32x4 z = {0u, 0u, 0u, 0u};
            #pragma unroll
            for (int i = 0; i < 4; ++i) d[sub * 4 + i] = z;
        }
    }

    const int wave  = tid >> 5;
    const int lane  = tid & 31;
    const int m     = lane & 15;
    const int s     = lane >> 4;
    const int Mtile = wave & 1;   // pixels 0-15 / 16-31
    const int Ntile = wave >> 1;  // co group of 16

    const int pixA = Mtile * 16 + m;     // A-row pixel
    const int py   = pixA >> 3, px = pixA & 7;

    v8f acc = {0.f, 0.f, 0.f, 0.f, 0.f, 0.f, 0.f, 0.f};

    for (int tap = 0; tap < 9; ++tap) {
        const int ky = tap / 3, kx = tap % 3;
        __syncthreads();   // previous tap's wt fully consumed
        {   // 16KB slab copy of pre-converted bf16 weights
            const u32x4* src = (const u32x4*)&wtap[tap * 8192];
            u32x4*       dst = (u32x4*)wt;
            #pragma unroll
            for (int i = 0; i < 4; ++i) dst[tid + 256 * i] = src[tid + 256 * i];
        }
        __syncthreads();

        const unsigned short* arow = &acts[((py + ky) * 10 + (px + kx)) * 128];
        #pragma unroll
        for (int kk = 0; kk < 4; ++kk) {
            BFv am;
            const unsigned short* ab = arow + kk * 32 + s * 16;
            am.q[0] = *(const u32x4*)ab;
            am.q[1] = *(const u32x4*)(ab + 8);
            BFv bm;
            const unsigned short* bp = &wt[(Ntile * 16 + m) * 128 + kk * 32 + s * 16];
            bm.q[0] = *(const u32x4*)bp;
            bm.q[1] = *(const u32x4*)(bp + 8);
            acc = wmma_bf16(am, bm, acc);
        }
    }

    // ---- stage D into LDS, then coalesced residual store ------------------
    __syncthreads();
    #pragma unroll
    for (int r = 0; r < 8; ++r) {
        int co  = Ntile * 16 + m;
        int pix = Mtile * 16 + r + s * 8;
        outT[co * 32 + pix] = acc[r];
    }
    __syncthreads();
    for (int e = tid; e < 2048; e += 256) {
        int co = e >> 5, pix = e & 31;
        int gy = Y0 + (pix >> 3), gx = X0 + (pix & 7);
        size_t idx = (((size_t)b * 64 + co) * 256 + gy) * 256 + gx;
        out[idx] = qin[idx] + cb[co] + outT[e];
    }
}

extern "C" void kernel_launch(void* const* d_in, const int* in_sizes, int n_in,
                              void* d_out, int out_size, void* d_ws, size_t ws_size,
                              hipStream_t stream) {
    const float* q   = (const float*)d_in[0];
    const float* v   = (const float*)d_in[1];
    const float* qW  = (const float*)d_in[2];
    const float* qb  = (const float*)d_in[3];
    const float* vW  = (const float*)d_in[4];
    const float* vb  = (const float*)d_in[5];
    const float* Km  = (const float*)d_in[6];
    const float* qng = (const float*)d_in[7];
    const float* qnb = (const float*)d_in[8];
    const float* vng = (const float*)d_in[9];
    const float* vnb = (const float*)d_in[10];
    const float* cW  = (const float*)d_in[11];
    const float* cb  = (const float*)d_in[12];
    float* out = (float*)d_out;

    unsigned short* spat = (unsigned short*)d_ws;
    unsigned short* wtap = spat + SPAT_ELEMS;
    unsigned short* qWbf = wtap + WTAP_ELEMS;
    unsigned short* Kbf  = qWbf + QW_ELEMS;

    prep_weights<<<(WTAP_ELEMS + 255) / 256, 256, 0, stream>>>(cW, qW, Km,
                                                               wtap, qWbf, Kbf);
    attn_stage1<<<8192, 128, 0, stream>>>(q, v, qWbf, qb, vW, vb, Kbf,
                                          qng, qnb, vng, vnb, spat);
    conv_stage2<<<16384, 256, 0, stream>>>(spat, wtap, cb, q, out);
}
